// TargetEncoder_4063039062792
// MI455X (gfx1250) — matrix operand order repaired and checked by reading, not compiled, not run on hardware
//
#include <hip/hip_runtime.h>
#include <hip/hip_bf16.h>

// ---------------------------------------------------------------------------
// TargetEncoder for MI455X (gfx1250, wave32, WMMA).
// Dense layers: v_wmma_f32_16x16x32_f16, one wave = 16x64 output tile.
// Weights are pre-packed into the WMMA B-fragment layout in f16, so the
// GEMM inner loop is: 4x b128 A loads + 8 cvt_pk, one 8-load clause for the
// four B fragments, then 4 back-to-back wmmas with partial loadcnt waits.
// Irregular graph work (frames, KNN, edge geometry, attention, LN) runs on
// wave32 kernels with LDS + shfl reductions.
// ---------------------------------------------------------------------------

#define NN        8192
#define A_IN      14
#define NAUG      10
#define NATM      15          // 5 + 10
#define LOCALD    128
#define PAIRD     64
#define KNBR      32
#define NHEADS    8
#define HDIM      16
#define NRBF      16
#define RCLAMP    32
#define NFEAT     302         // 3*15 + 16*15 + 15 + 2
#define FEAT_PAD  320
#define NEDGE     (NN*KNBR)   // 262144
#define GEMM_WAVES 4
#define NT         4          // 16x(16*NT) tile per wave

typedef __attribute__((ext_vector_type(16))) _Float16 v16h;
typedef __attribute__((ext_vector_type(8)))  float    v8f;

// ---------------------------------------------------------------------------
// helpers
// ---------------------------------------------------------------------------
__device__ __forceinline__ float gelu_f(float x) {
  const float c = 0.7978845608028654f;
  return 0.5f * x * (1.0f + tanhf(c * (x + 0.044715f * x * x * x)));
}
__device__ __forceinline__ float wred_sum(float v) {
  #pragma unroll
  for (int m = 16; m > 0; m >>= 1) v += __shfl_xor(v, m, 32);
  return v;
}
__device__ __forceinline__ float wred_max(float v) {
  #pragma unroll
  for (int m = 16; m > 0; m >>= 1) v = fmaxf(v, __shfl_xor(v, m, 32));
  return v;
}
__device__ __forceinline__ float rbf_val(float d, int b) {
  const float sigma = 22.0f / 16.0f;              // (d_max-d_min)/bins
  float c = 22.0f * (float)b / 15.0f;             // linspace(0,22,16)
  float z = (d - c) / sigma;
  return __expf(-z * z);
}

// ---------------------------------------------------------------------------
// Kernel 1: frames, local coords, augmentation, node features (one thread/node)
// ---------------------------------------------------------------------------
__global__ void frames_kernel(const float* __restrict__ pos,
                              const int* __restrict__ is_target,
                              const int* __restrict__ hotspots,
                              const int* __restrict__ maski,
                              const float* __restrict__ aug_w,
                              float* __restrict__ Rb,   // [N,9]
                              float* __restrict__ tb,   // [N,3]
                              float* __restrict__ qp,   // [N,3]  positions[:,4]
                              float* __restrict__ pos15,// [N,45]
                              float* __restrict__ lp64, // [N,64] zero-padded lp
                              float* __restrict__ feats,// [N,320] zero-padded
                              float* __restrict__ maskf,
                              float* __restrict__ istf) {
  int n = blockIdx.x * blockDim.x + threadIdx.x;
  if (n >= NN) return;
  float p[A_IN][3];
  #pragma unroll
  for (int a = 0; a < A_IN; ++a)
    #pragma unroll
    for (int d = 0; d < 3; ++d)
      p[a][d] = pos[(size_t)n * A_IN * 3 + a * 3 + d];

  float nv[3], ca[3], cv[3];
  #pragma unroll
  for (int d = 0; d < 3; ++d) { nv[d] = p[0][d]; ca[d] = p[1][d]; cv[d] = p[2][d]; }

  float e1[3], e2[3], e3[3];
  {
    float v[3]; float s = 0.f;
    #pragma unroll
    for (int d = 0; d < 3; ++d) { v[d] = cv[d] - ca[d]; s += v[d] * v[d]; }
    float inv = 1.0f / (sqrtf(s) + 1e-8f);
    #pragma unroll
    for (int d = 0; d < 3; ++d) e1[d] = v[d] * inv;
  }
  {
    float v2[3]; float dp = 0.f;
    #pragma unroll
    for (int d = 0; d < 3; ++d) v2[d] = nv[d] - ca[d];
    #pragma unroll
    for (int d = 0; d < 3; ++d) dp += v2[d] * e1[d];
    float w[3]; float s = 0.f;
    #pragma unroll
    for (int d = 0; d < 3; ++d) { w[d] = v2[d] - dp * e1[d]; s += w[d] * w[d]; }
    float inv = 1.0f / (sqrtf(s) + 1e-8f);
    #pragma unroll
    for (int d = 0; d < 3; ++d) e2[d] = w[d] * inv;
  }
  e3[0] = e1[1] * e2[2] - e1[2] * e2[1];
  e3[1] = e1[2] * e2[0] - e1[0] * e2[2];
  e3[2] = e1[0] * e2[1] - e1[1] * e2[0];

  float R[9]; // R[d*3+e], columns e1,e2,e3
  #pragma unroll
  for (int d = 0; d < 3; ++d) { R[d*3+0] = e1[d]; R[d*3+1] = e2[d]; R[d*3+2] = e3[d]; }

  float loc[A_IN][3];
  #pragma unroll
  for (int a = 0; a < A_IN; ++a)
    #pragma unroll
    for (int e = 0; e < 3; ++e) {
      float s = 0.f;
      #pragma unroll
      for (int d = 0; d < 3; ++d) s += (p[a][d] - ca[d]) * R[d*3+e];
      loc[a][e] = s;
    }

  float aug[NAUG][3]; float ssum = 0.f;
  #pragma unroll
  for (int k = 0; k < NAUG; ++k) {
    #pragma unroll
    for (int d = 0; d < 3; ++d) {
      float s = 0.f;
      #pragma unroll
      for (int a = 0; a < A_IN; ++a) s += loc[a][d] * aug_w[a * NAUG + k];
      aug[k][d] = s; ssum += s * s;
    }
  }
  float rms = sqrtf(ssum / (float)NAUG);
  float ainv = 1.0f / (rms + 1e-6f);

  float lp[NATM][3];
  #pragma unroll
  for (int a = 0; a < 5; ++a)
    #pragma unroll
    for (int e = 0; e < 3; ++e) lp[a][e] = loc[a][e];
  #pragma unroll
  for (int k = 0; k < NAUG; ++k)
    #pragma unroll
    for (int e = 0; e < 3; ++e) lp[5 + k][e] = aug[k][e] * ainv;

  // positions = lp @ R^T + t   (lp == to_local(positions) since R orthonormal)
  #pragma unroll
  for (int a = 0; a < NATM; ++a)
    #pragma unroll
    for (int d = 0; d < 3; ++d) {
      float s = 0.f;
      #pragma unroll
      for (int e = 0; e < 3; ++e) s += lp[a][e] * R[d*3+e];
      pos15[(size_t)n * 45 + a * 3 + d] = s + ca[d];
    }
  #pragma unroll
  for (int d = 0; d < 3; ++d) {
    qp[n * 3 + d] = pos15[(size_t)n * 45 + 4 * 3 + d];
    tb[n * 3 + d] = ca[d];
  }
  #pragma unroll
  for (int i = 0; i < 9; ++i) Rb[n * 9 + i] = R[i];

  float ist = (is_target[n] > 0) ? 1.f : 0.f;
  float hot = (hotspots[n] > 0) ? 1.f : 0.f;
  maskf[n] = ((maski[n] > 0) ? 1.f : 0.f) * ist;
  istf[n] = ist;

  float* f = feats + (size_t)n * FEAT_PAD;
  float dist[NATM];
  #pragma unroll
  for (int a = 0; a < NATM; ++a) {
    float s = 0.f;
    #pragma unroll
    for (int e = 0; e < 3; ++e) { s += lp[a][e] * lp[a][e]; lp64[(size_t)n * 64 + a * 3 + e] = lp[a][e]; }
    dist[a] = sqrtf(s);
  }
  #pragma unroll
  for (int i = 45; i < 64; ++i) lp64[(size_t)n * 64 + i] = 0.f;
  int off = 0;
  #pragma unroll
  for (int a = 0; a < NATM; ++a) {
    float inv = 1.0f / (dist[a] + 1e-8f);
    #pragma unroll
    for (int e = 0; e < 3; ++e) f[off++] = lp[a][e] * inv;
  }
  #pragma unroll
  for (int a = 0; a < NATM; ++a)
    #pragma unroll
    for (int b = 0; b < NRBF; ++b) f[off++] = rbf_val(dist[a], b);
  #pragma unroll
  for (int a = 0; a < NATM; ++a) f[off++] = logf(dist[a] + 1.0f);
  f[off++] = ist;
  f[off++] = hot;
  for (; off < FEAT_PAD; ++off) f[off] = 0.f;
}

// ---------------------------------------------------------------------------
// Kernel 2: brute-force KNN, one 128-thread block per query node.
// ---------------------------------------------------------------------------
__global__ void knn_kernel(const float* __restrict__ qp,
                           const int* __restrict__ batch,
                           const float* __restrict__ maskf,
                           int* __restrict__ nbr) {
  int n = blockIdx.x;
  int tid = threadIdx.x;             // 0..127
  __shared__ float sd[128 * KNBR];
  __shared__ int   si[128 * KNBR];
  float* md = sd + tid * KNBR;
  int*   mi = si + tid * KNBR;
  for (int i = 0; i < KNBR; ++i) { md[i] = 3.0e38f; mi[i] = -1; }

  float qx = qp[n*3], qy = qp[n*3+1], qz = qp[n*3+2];
  int bn = batch[n];
  if (maskf[n] > 0.f) {
    for (int j = tid; j < NN; j += 128) {
      if (batch[j] != bn || maskf[j] <= 0.f) continue;
      float dx = qp[j*3]   - qx;
      float dy = qp[j*3+1] - qy;
      float dz = qp[j*3+2] - qz;
      float d2 = dx*dx + dy*dy + dz*dz;
      if (d2 < md[KNBR-1]) {
        int pp = KNBR - 1;
        while (pp > 0 && md[pp-1] > d2) { md[pp] = md[pp-1]; mi[pp] = mi[pp-1]; --pp; }
        md[pp] = d2; mi[pp] = j;
      }
    }
  }
  __syncthreads();
  if (tid == 0) {
    int head[128];
    for (int l = 0; l < 128; ++l) head[l] = 0;
    for (int s = 0; s < KNBR; ++s) {
      float best = 3.0e38f; int bl = -1;
      for (int l = 0; l < 128; ++l) {
        int h = head[l];
        if (h < KNBR) { float v = sd[l*KNBR + h]; if (v < best) { best = v; bl = l; } }
      }
      int idx = -1;
      if (bl >= 0 && best < 1.0e37f) { idx = si[bl*KNBR + head[bl]]; head[bl]++; }
      nbr[(size_t)n * KNBR + s] = idx;
    }
  }
}

// ---------------------------------------------------------------------------
// Kernel 3a: pack a weight matrix B [K,N] f32 into the WMMA B-fragment f16
// layout. Element ((g*Kb + kb)*32 + lane)*16 + j = B[kb*32+(lane>>4)*16+j]
// [g*16+(lane&15)], with rows >= Krows zero-filled (absorbs K padding).
// ---------------------------------------------------------------------------
__global__ void pack_b_kernel(const float* __restrict__ B,
                              _Float16* __restrict__ Bp,
                              int Krows, int Kpad, int Nn, int ldb) {
  int i = blockIdx.x * blockDim.x + threadIdx.x;
  int kblocks = Kpad >> 5;
  int total = (Nn >> 4) * kblocks * 32;
  if (i >= total) return;
  int lane = i & 31;
  int kb = (i >> 5) % kblocks;
  int g  = (i >> 5) / kblocks;
  int idx = lane & 15, half = lane >> 4;
  int col = g * 16 + idx;
  int krow = kb * 32 + half * 16;
  _Float16* dst = Bp + ((size_t)i << 4);
  #pragma unroll
  for (int j = 0; j < 16; ++j) {
    int k = krow + j;
    dst[j] = (_Float16)((k < Krows) ? B[(size_t)k * ldb + col] : 0.f);
  }
}

// ---------------------------------------------------------------------------
// Kernel 3b: WMMA GEMM, guard-free.  D = act(A@Bp + Cinit + bias).
// REQUIREMENTS: M%16==0, Nn%64==0, Kd%32==0, lda%4==0, A 16B-aligned.
// A [M,lda] f32 row-major; Bp packed by pack_b_kernel. One wave = 16x64 tile
// (4 accumulators, A fragment reused 4x). act: 0=linear, 1=gelu.
// ---------------------------------------------------------------------------
__global__ void gemm_wmma_kernel(const float* __restrict__ A,
                                 const _Float16* __restrict__ Bp,
                                 const float* __restrict__ Cinit,
                                 float* __restrict__ D,
                                 const float* __restrict__ bias,
                                 int M, int Nn, int Kd,
                                 int lda, int ldc, int act) {
  int ntn = Nn >> 6;                           // tiles of 64 cols
  int ntm = M >> 4;
  int tile = blockIdx.x * GEMM_WAVES + (threadIdx.x >> 5);
  if (tile >= ntm * ntn) return;               // whole-wave uniform exit
  int tm = (tile / ntn) << 4;
  int tn = (tile % ntn) << 6;
  int lane = threadIdx.x & 31;
  int half = lane >> 4;                        // 0/1
  int idx  = lane & 15;
  int kblocks = Kd >> 5;

  v8f acc[NT];
  {
    v8f z = {};
    #pragma unroll
    for (int t = 0; t < NT; ++t) acc[t] = z;
  }
  if (Cinit) {
    #pragma unroll
    for (int t = 0; t < NT; ++t)
      #pragma unroll
      for (int r = 0; r < 8; ++r)
        acc[t][r] = Cinit[(size_t)(tm + r + 8 * half) * ldc + tn + t * 16 + idx];
  }

  const float* Arow = A + (size_t)(tm + idx) * lda;   // this lane's A row
  // per-column-group packed-B fragment base (v16h units)
  const v16h* bfrag[NT];
  #pragma unroll
  for (int t = 0; t < NT; ++t) {
    int g = (tn >> 4) + t;
    bfrag[t] = (const v16h*)Bp + (size_t)g * kblocks * 32 + lane;
  }

  for (int kb = 0; kb < kblocks; ++kb) {
    int kk = kb << 5;
    if (kb + 1 < kblocks)
      __builtin_prefetch(Arow + kk + 32, 0, 1);       // CDNA5 global_prefetch_b8

    // A 16x32 f16 fragment (contiguous order, matches packed B):
    // halves 0..15 : K = kk + half*16 + j
    float4 a0 = *(const float4*)(Arow + kk + half * 16);
    float4 a1 = *(const float4*)(Arow + kk + half * 16 + 4);
    float4 a2 = *(const float4*)(Arow + kk + half * 16 + 8);
    float4 a3 = *(const float4*)(Arow + kk + half * 16 + 12);
    v16h fa;
    fa[0]=(_Float16)a0.x;  fa[1]=(_Float16)a0.y;  fa[2]=(_Float16)a0.z;  fa[3]=(_Float16)a0.w;
    fa[4]=(_Float16)a1.x;  fa[5]=(_Float16)a1.y;  fa[6]=(_Float16)a1.z;  fa[7]=(_Float16)a1.w;
    fa[8]=(_Float16)a2.x;  fa[9]=(_Float16)a2.y;  fa[10]=(_Float16)a2.z; fa[11]=(_Float16)a2.w;
    fa[12]=(_Float16)a3.x; fa[13]=(_Float16)a3.y; fa[14]=(_Float16)a3.z; fa[15]=(_Float16)a3.w;

    // issue all four B-fragment loads first (one clause), then the four
    // wmmas: lets the scheduler use partial loadcnt waits and overlap
    // B-fragment latency with the matrix pipe.
    v16h fb[NT];
    #pragma unroll
    for (int t = 0; t < NT; ++t) fb[t] = bfrag[t][(size_t)kb * 32];
    #pragma unroll
    for (int t = 0; t < NT; ++t)
      acc[t] = __builtin_amdgcn_wmma_f32_16x16x32_f16(
          false, fa, false, fb[t], (short)0, acc[t], false, false);
  }

  #pragma unroll
  for (int t = 0; t < NT; ++t) {
    int nc = tn + t * 16 + idx;
    float bv = bias ? bias[nc] : 0.f;          // hoisted: one load per tile col
    #pragma unroll
    for (int r = 0; r < 8; ++r) {
      float x = acc[t][r] + bv;
      if (act == 1) x = gelu_f(x);
      D[(size_t)(tm + r + 8 * half) * ldc + nc] = x;
    }
  }
}

// ---------------------------------------------------------------------------
// Kernel 4: per-edge geometric features (block-invariant; computed once).
// ef layout [E,128]: [0:16) rbf(d_ca), [16:61) dir, [61:70) rot, [70:115) vec,
// [115:128) zero pad.
// ---------------------------------------------------------------------------
__global__ void edge_feat_kernel(const int* __restrict__ nbr,
                                 const int* __restrict__ resi,
                                 const int* __restrict__ chain,
                                 const int* __restrict__ batch,
                                 const float* __restrict__ maskf,
                                 const float* __restrict__ Rb,
                                 const float* __restrict__ tb,
                                 const float* __restrict__ pos15,
                                 float* __restrict__ ef,
                                 float* __restrict__ pm,
                                 float* __restrict__ samef,
                                 int* __restrict__ relb) {
  int e = blockIdx.x * blockDim.x + threadIdx.x;
  if (e >= NEDGE) return;
  int n = e / KNBR;
  int j = nbr[e];
  int js = (j < 0) ? 0 : j;
  float pmv = maskf[n] * maskf[js] * ((j != -1) ? 1.f : 0.f);
  bool same = (chain[n] == chain[js]) && (batch[n] == batch[js]);
  int rel = resi[js] - resi[n];
  rel = (rel < -RCLAMP) ? -RCLAMP : ((rel > RCLAMP) ? RCLAMP : rel);
  rel += RCLAMP;

  float Rn[9], Rj[9], tn3[3], tj3[3];
  #pragma unroll
  for (int i = 0; i < 9; ++i) { Rn[i] = Rb[n*9+i]; Rj[i] = Rb[js*9+i]; }
  #pragma unroll
  for (int d = 0; d < 3; ++d) { tn3[d] = tb[n*3+d]; tj3[d] = tb[js*3+d]; }

  float* o = ef + (size_t)e * 128;
  float dca;
  {
    float s = 0.f;
    #pragma unroll
    for (int d = 0; d < 3; ++d) { float dd = tj3[d] - tn3[d]; s += dd * dd; }
    dca = sqrtf(s);
  }
  #pragma unroll
  for (int b = 0; b < NRBF; ++b) o[b] = rbf_val(dca, b);

  #pragma unroll
  for (int a = 0; a < NATM; ++a) {
    float rv[3];
    #pragma unroll
    for (int ee = 0; ee < 3; ++ee) {
      float s = 0.f;
      #pragma unroll
      for (int d = 0; d < 3; ++d)
        s += (pos15[(size_t)js * 45 + a * 3 + d] - tn3[d]) * Rn[d*3+ee];
      rv[ee] = s;
    }
    float nrm = sqrtf(rv[0]*rv[0] + rv[1]*rv[1] + rv[2]*rv[2]);
    float inv = 1.0f / (nrm + 1e-8f);
    #pragma unroll
    for (int ee = 0; ee < 3; ++ee) {
      o[16 + a * 3 + ee] = rv[ee] * inv;   // dir_feat
      o[70 + a * 3 + ee] = rv[ee];         // vec_feat
    }
  }
  #pragma unroll
  for (int i = 0; i < 3; ++i)
    #pragma unroll
    for (int jj = 0; jj < 3; ++jj) {
      float s = 0.f;
      #pragma unroll
      for (int d = 0; d < 3; ++d) s += Rn[d*3+i] * Rj[d*3+jj];
      o[61 + i * 3 + jj] = s;              // rot_feat
    }
  #pragma unroll
  for (int i = 115; i < 128; ++i) o[i] = 0.f;

  pm[e] = pmv;
  samef[e] = same ? 1.f : 0.f;
  relb[e] = rel;
}

// Assemble concatenated pair-projection weight [128,64] (per block's params)
__global__ void wcat_kernel(const float* __restrict__ w_dist,
                            const float* __restrict__ w_dir,
                            const float* __restrict__ w_rot,
                            const float* __restrict__ w_vec,
                            float* __restrict__ Wcat) {
  int i = blockIdx.x * blockDim.x + threadIdx.x;
  if (i >= 128 * 64) return;
  int r = i >> 6, c = i & 63;
  float v = 0.f;
  if (r < 16)       v = w_dist[r * 64 + c];
  else if (r < 61)  v = w_dir[(r - 16) * 64 + c];
  else if (r < 70)  v = w_rot[(r - 61) * 64 + c];
  else if (r < 115) v = w_vec[(r - 70) * 64 + c];
  Wcat[i] = v;
}

// pair_init[e,:] = same ? w_relpos[rel,:] : 0   (relpos one-hot @ w_relpos)
__global__ void pair_base_kernel(const float* __restrict__ w_relpos,
                                 const int* __restrict__ relb,
                                 const float* __restrict__ samef,
                                 float* __restrict__ pair) {
  size_t i = (size_t)blockIdx.x * blockDim.x + threadIdx.x;
  if (i >= (size_t)NEDGE * PAIRD) return;
  size_t e = i >> 6;
  int c = (int)(i & 63);
  pair[i] = samef[e] * w_relpos[(size_t)relb[e] * PAIRD + c];
}

// ---------------------------------------------------------------------------
// LayerNorm (one wave per row; W in {64,128}); out2 optional copy (inc init).
// ---------------------------------------------------------------------------
__global__ void ln_kernel(const float* __restrict__ x,
                          float* __restrict__ out, float* __restrict__ out2,
                          const float* __restrict__ s, const float* __restrict__ b,
                          int W) {
  int row = blockIdx.x;
  int lane = threadIdx.x;
  const float* xr = x + (size_t)row * W;
  float v[4]; float sum = 0.f, sq = 0.f;
  #pragma unroll
  for (int i = 0; i < 4; ++i) {
    int c = lane + i * 32;
    float t = (c < W) ? xr[c] : 0.f;
    v[i] = t; sum += t; sq += t * t;
  }
  sum = wred_sum(sum); sq = wred_sum(sq);
  float mean = sum / (float)W;
  float var = sq / (float)W - mean * mean;
  float inv = rsqrtf(var + 1e-5f);
  #pragma unroll
  for (int i = 0; i < 4; ++i) {
    int c = lane + i * 32;
    if (c < W) {
      float y = (v[i] - mean) * inv * s[c] + b[c];
      out[(size_t)row * W + c] = y;
      if (out2) out2[(size_t)row * W + c] = y;
    }
  }
}

// inc += upd*mask; local = LN(local + upd*mask)   (W=128, one wave per row)
__global__ void resid_ln_kernel(float* __restrict__ local,
                                const float* __restrict__ upd,
                                float* __restrict__ inc,
                                const float* __restrict__ maskf,
                                const float* __restrict__ s,
                                const float* __restrict__ b) {
  int row = blockIdx.x;
  int lane = threadIdx.x;
  float m = maskf ? maskf[row] : 1.f;
  float v[4]; float sum = 0.f, sq = 0.f;
  #pragma unroll
  for (int i = 0; i < 4; ++i) {
    size_t p = (size_t)row * LOCALD + lane + i * 32;
    float u = upd[p] * m;
    inc[p] += u;
    float x = local[p] + u;
    v[i] = x; sum += x; sq += x * x;
  }
  sum = wred_sum(sum); sq = wred_sum(sq);
  float mean = sum / (float)LOCALD;
  float var = sq / (float)LOCALD - mean * mean;
  float inv = rsqrtf(var + 1e-5f);
  #pragma unroll
  for (int i = 0; i < 4; ++i) {
    int c = lane + i * 32;
    local[(size_t)row * LOCALD + c] = (v[i] - mean) * inv * s[c] + b[c];
  }
}

// ---------------------------------------------------------------------------
// Attention over K=32 neighbors, 8 heads x 16 dims. One wave per node.
// ---------------------------------------------------------------------------
__global__ void attn_kernel(const float* __restrict__ qb,
                            const float* __restrict__ kb,
                            const float* __restrict__ vb,
                            const float* __restrict__ pair2,
                            const float* __restrict__ w_bias, // [64,8]
                            const float* __restrict__ pm,
                            const int* __restrict__ nbr,
                            float* __restrict__ ob) {
  int n = blockIdx.x;
  int lane = threadIdx.x;                      // 0..31, one neighbor slot
  __shared__ float qs[LOCALD];
  __shared__ float attn_s[NHEADS * KNBR];
  __shared__ int   js_s[KNBR];
  __shared__ float wb_s[PAIRD * NHEADS];
  for (int i = lane; i < LOCALD; i += 32) qs[i] = qb[(size_t)n * LOCALD + i];
  for (int i = lane; i < PAIRD * NHEADS; i += 32) wb_s[i] = w_bias[i];
  size_t e = (size_t)n * KNBR + lane;
  int j = nbr[e];
  int js = (j < 0) ? 0 : j;
  js_s[lane] = js;
  float pmv = pm[e];
  __syncthreads();

  float bias[NHEADS];
  #pragma unroll
  for (int h = 0; h < NHEADS; ++h) bias[h] = 0.f;
  for (int p = 0; p < PAIRD; ++p) {
    float pv = pair2[e * PAIRD + p];
    #pragma unroll
    for (int h = 0; h < NHEADS; ++h) bias[h] += pv * wb_s[p * NHEADS + h];
  }
  #pragma unroll
  for (int h = 0; h < NHEADS; ++h) {
    float dot = 0.f;
    #pragma unroll
    for (int d = 0; d < HDIM; ++d)
      dot += qs[h * HDIM + d] * kb[(size_t)js * LOCALD + h * HDIM + d];
    float l = dot * 0.25f + bias[h];          // 1/sqrt(16)
    l = (pmv > 0.f) ? l : -1e9f;
    float mx = wred_max(l);
    float ex = __expf(l - mx);
    float ssum = wred_sum(ex);
    attn_s[h * KNBR + lane] = ex / ssum;
  }
  __syncthreads();
  #pragma unroll
  for (int i = 0; i < 4; ++i) {
    int c = lane * 4 + i;
    int h = c >> 4;
    float acc = 0.f;
    for (int kk = 0; kk < KNBR; ++kk)
      acc += attn_s[h * KNBR + kk] * vb[(size_t)js_s[kk] * LOCALD + c];
    ob[(size_t)n * LOCALD + c] = acc;
  }
}

// elementwise a *= b
__global__ void ewise_mul_kernel(float* __restrict__ a, const float* __restrict__ b, size_t n) {
  size_t i = (size_t)blockIdx.x * blockDim.x + threadIdx.x;
  if (i < n) a[i] *= b[i];
}

// out = is_t>0 ? local + LN(inc)*s+b : 0
__global__ void final_kernel(const float* __restrict__ local,
                             const float* __restrict__ inc,
                             const float* __restrict__ s,
                             const float* __restrict__ b,
                             const float* __restrict__ istf,
                             float* __restrict__ out) {
  int row = blockIdx.x;
  int lane = threadIdx.x;
  const float* xr = inc + (size_t)row * LOCALD;
  float v[4]; float sum = 0.f, sq = 0.f;
  #pragma unroll
  for (int i = 0; i < 4; ++i) {
    float t = xr[lane + i * 32];
    v[i] = t; sum += t; sq += t * t;
  }
  sum = wred_sum(sum); sq = wred_sum(sq);
  float mean = sum / (float)LOCALD;
  float var = sq / (float)LOCALD - mean * mean;
  float inv = rsqrtf(var + 1e-5f);
  float g = (istf[row] > 0.f) ? 1.f : 0.f;
  #pragma unroll
  for (int i = 0; i < 4; ++i) {
    int c = lane + i * 32;
    float y = (v[i] - mean) * inv * s[c] + b[c] + local[(size_t)row * LOCALD + c];
    out[(size_t)row * LOCALD + c] = y * g;
  }
}

// ---------------------------------------------------------------------------
// host side
// ---------------------------------------------------------------------------
static void packB(hipStream_t st, const float* B, _Float16* Bp,
                  int Krows, int Kpad, int Nn, int ldb) {
  int total = (Nn >> 4) * (Kpad >> 5) * 32;
  pack_b_kernel<<<(total + 255) / 256, 256, 0, st>>>(B, Bp, Krows, Kpad, Nn, ldb);
}
static void gemm(hipStream_t st, const float* A, const _Float16* Bp, const float* Ci,
                 float* D, const float* bias, int M, int Nn, int Kd,
                 int lda, int ldc, int act) {
  int tiles = (M >> 4) * (Nn >> 6);
  int grid = (tiles + GEMM_WAVES - 1) / GEMM_WAVES;
  gemm_wmma_kernel<<<grid, 32 * GEMM_WAVES, 0, st>>>(A, Bp, Ci, D, bias,
                                                     M, Nn, Kd, lda, ldc, act);
}

// params flattened in jax pytree (sorted-key) order after the 7 array inputs
enum { B_OUT = 0, B_BO, B_LN1B, B_LN1S, B_LN2B, B_LN2S, B_PAIRB1, B_PAIRB2,
       B_PAIRLNB, B_PAIRLNS, B_PAIRW1, B_PAIRW2, B_POSB1, B_POSB2, B_POSW1,
       B_POSW2, B_WBIAS, B_WDIR, B_WDIST, B_WGATE, B_WOUT, B_WRELPOS, B_WROT,
       B_WUP, B_WVEC, B_WK, B_WO, B_WQ, B_WV, NBP };
#define PI_AUGW   7
#define PI_BLK0   8
#define PI_FEATLNB (PI_BLK0 + 3 * NBP + 0)
#define PI_FEATLNS (PI_BLK0 + 3 * NBP + 1)
#define PI_FEATW1  (PI_BLK0 + 3 * NBP + 2)
#define PI_FEATW2  (PI_BLK0 + 3 * NBP + 3)
#define PI_FINLNB  (PI_BLK0 + 3 * NBP + 4)
#define PI_FINLNS  (PI_BLK0 + 3 * NBP + 5)

extern "C" void kernel_launch(void* const* d_in, const int* in_sizes, int n_in,
                              void* d_out, int out_size, void* d_ws, size_t ws_size,
                              hipStream_t stream) {
  (void)in_sizes; (void)n_in; (void)out_size; (void)ws_size;

  const float* pos  = (const float*)d_in[0];
  const int*   resi = (const int*)d_in[1];
  const int*   chn  = (const int*)d_in[2];
  const int*   bat  = (const int*)d_in[3];
  const int*   ist  = (const int*)d_in[4];
  const int*   hot  = (const int*)d_in[5];
  const int*   msk  = (const int*)d_in[6];
  float* out = (float*)d_out;

  auto prm = [&](int i) { return (const float*)d_in[i]; };
  auto bpp = [&](int blk, int which) { return (const float*)d_in[PI_BLK0 + blk * NBP + which]; };

  // ---- workspace carve-up (floats; every size is a multiple of 4 so all
  //      buffers stay 16-byte aligned for float4 / v16h loads) ----
  float* W0 = (float*)d_ws;
  size_t off = 0;
  auto alloc = [&](size_t nf) { float* p = W0 + off; off += nf; return p; };
  float* feats  = alloc((size_t)NN * FEAT_PAD);
  float* Rb     = alloc((size_t)NN * 9);
  float* tb     = alloc((size_t)NN * 3);
  float* qp     = alloc((size_t)NN * 3);
  float* pos15  = alloc((size_t)NN * 45);
  float* lp64   = alloc((size_t)NN * 64);
  float* maskf  = alloc(NN);
  float* istf   = alloc(NN);
  float* localb = alloc((size_t)NN * LOCALD);
  float* incb   = alloc((size_t)NN * LOCALD);
  float* updb   = alloc((size_t)NN * LOCALD);
  float* obuf   = alloc((size_t)NN * LOCALD);
  float* qbuf   = alloc((size_t)NN * LOCALD);
  float* kbuf   = alloc((size_t)NN * LOCALD);
  float* vbuf   = alloc((size_t)NN * LOCALD);
  float* hbuf   = alloc((size_t)NN * LOCALD);
  float* t512   = alloc((size_t)NN * 512);
  float* g256   = alloc((size_t)NN * 256);
  float* u256   = alloc((size_t)NN * 256);
  float* Wcat   = alloc(128 * 64);
  _Float16* Bp  = (_Float16*)alloc((size_t)FEAT_PAD * 512 / 2); // packed-B scratch (max 320x512 halves)
  float* pairb  = alloc((size_t)NEDGE * PAIRD);
  float* pair2b = alloc((size_t)NEDGE * PAIRD);
  float* efb    = alloc((size_t)NEDGE * 128);      // block-invariant edge feats
  float* ehid   = alloc((size_t)NEDGE * 128);      // pair-MLP hidden
  float* pmb    = alloc(NEDGE);
  float* samefb = alloc(NEDGE);
  int* relb = (int*)(W0 + off); off += NEDGE;
  int* nbrb = (int*)(W0 + off); off += NEDGE;

  // ---- 1. frames + node features ----
  frames_kernel<<<(NN + 127) / 128, 128, 0, stream>>>(
      pos, ist, hot, msk, prm(PI_AUGW),
      Rb, tb, qp, pos15, lp64, feats, maskf, istf);

  // ---- 2. KNN graph ----
  knn_kernel<<<NN, 128, 0, stream>>>(qp, bat, maskf, nbrb);

  // ---- 3. feature MLP: local = LN(gelu(feats@W1)@W2) ; inc = local ----
  packB(stream, prm(PI_FEATW1), Bp, NFEAT, FEAT_PAD, 512, 512);
  gemm(stream, feats, Bp, nullptr, t512, nullptr,
       NN, 512, FEAT_PAD, FEAT_PAD, 512, /*gelu*/1);
  packB(stream, prm(PI_FEATW2), Bp, 512, 512, LOCALD, LOCALD);
  gemm(stream, t512, Bp, nullptr, localb, nullptr,
       NN, LOCALD, 512, 512, LOCALD, 0);
  ln_kernel<<<NN, 32, 0, stream>>>(localb, localb, incb,
                                   prm(PI_FEATLNS), prm(PI_FEATLNB), LOCALD);

  // ---- 4. block-invariant edge features ----
  edge_feat_kernel<<<NEDGE / 256, 256, 0, stream>>>(
      nbrb, resi, chn, bat, maskf, Rb, tb, pos15, efb, pmb, samefb, relb);

  // ---- 5. transformer blocks ----
  for (int blk = 0; blk < 3; ++blk) {
    // pair = LN(relpos_gather + ef@Wcat) ; pair2 = mlp(pair)
    pair_base_kernel<<<(int)(((size_t)NEDGE * PAIRD) / 256), 256, 0, stream>>>(
        bpp(blk, B_WRELPOS), relb, samefb, pairb);
    wcat_kernel<<<(128 * 64 + 255) / 256, 256, 0, stream>>>(
        bpp(blk, B_WDIST), bpp(blk, B_WDIR), bpp(blk, B_WROT), bpp(blk, B_WVEC), Wcat);
    packB(stream, Wcat, Bp, 128, 128, PAIRD, PAIRD);
    gemm(stream, efb, Bp, pairb /*Cinit; safe element-wise alias*/, pairb, nullptr,
         NEDGE, PAIRD, 128, 128, PAIRD, 0);
    ln_kernel<<<NEDGE, 32, 0, stream>>>(pairb, pairb, nullptr,
                                        bpp(blk, B_PAIRLNS), bpp(blk, B_PAIRLNB), PAIRD);
    packB(stream, bpp(blk, B_PAIRW1), Bp, PAIRD, PAIRD, 128, 128);
    gemm(stream, pairb, Bp, nullptr, ehid, bpp(blk, B_PAIRB1),
         NEDGE, 128, PAIRD, PAIRD, 128, /*gelu*/1);
    packB(stream, bpp(blk, B_PAIRW2), Bp, 128, 128, PAIRD, PAIRD);
    gemm(stream, ehid, Bp, nullptr, pair2b, bpp(blk, B_PAIRB2),
         NEDGE, PAIRD, 128, 128, PAIRD, 0);

    // attention
    packB(stream, bpp(blk, B_WQ), Bp, LOCALD, LOCALD, LOCALD, LOCALD);
    gemm(stream, localb, Bp, nullptr, qbuf, nullptr,
         NN, LOCALD, LOCALD, LOCALD, LOCALD, 0);
    packB(stream, bpp(blk, B_WK), Bp, LOCALD, LOCALD, LOCALD, LOCALD);
    gemm(stream, localb, Bp, nullptr, kbuf, nullptr,
         NN, LOCALD, LOCALD, LOCALD, LOCALD, 0);
    packB(stream, bpp(blk, B_WV), Bp, LOCALD, LOCALD, LOCALD, LOCALD);
    gemm(stream, localb, Bp, nullptr, vbuf, nullptr,
         NN, LOCALD, LOCALD, LOCALD, LOCALD, 0);
    attn_kernel<<<NN, 32, 0, stream>>>(qbuf, kbuf, vbuf, pair2b,
                                       bpp(blk, B_WBIAS), pmb, nbrb, obuf);
    packB(stream, bpp(blk, B_WO), Bp, LOCALD, LOCALD, LOCALD, LOCALD);
    gemm(stream, obuf, Bp, nullptr, updb, bpp(blk, B_BO),
         NN, LOCALD, LOCALD, LOCALD, LOCALD, 0);
    resid_ln_kernel<<<NN, 32, 0, stream>>>(localb, updb, incb, maskf,
                                           bpp(blk, B_LN1S), bpp(blk, B_LN1B));

    // h = local + mlp(lp_flat); gated FFN
    packB(stream, bpp(blk, B_POSW1), Bp, 45, 64, 256, 256);
    gemm(stream, lp64, Bp, nullptr, t512, bpp(blk, B_POSB1),
         NN, 256, 64, 64, 256, /*gelu*/1);
    packB(stream, bpp(blk, B_POSW2), Bp, 256, 256, LOCALD, LOCALD);
    gemm(stream, t512, Bp, localb /*Cinit = local*/, hbuf,
         bpp(blk, B_POSB2), NN, LOCALD, 256, 256, LOCALD, 0);
    packB(stream, bpp(blk, B_WGATE), Bp, LOCALD, LOCALD, 256, 256);
    gemm(stream, hbuf, Bp, nullptr, g256, nullptr,
         NN, 256, LOCALD, LOCALD, 256, /*gelu*/1);
    packB(stream, bpp(blk, B_WUP), Bp, LOCALD, LOCALD, 256, 256);
    gemm(stream, hbuf, Bp, nullptr, u256, nullptr,
         NN, 256, LOCALD, LOCALD, 256, 0);
    ewise_mul_kernel<<<(int)(((size_t)NN * 256 + 255) / 256), 256, 0, stream>>>(
        g256, u256, (size_t)NN * 256);
    packB(stream, bpp(blk, B_WOUT), Bp, 256, 256, LOCALD, LOCALD);
    gemm(stream, g256, Bp, nullptr, updb, bpp(blk, B_OUT),
         NN, LOCALD, 256, 256, LOCALD, 0);
    resid_ln_kernel<<<NN, 32, 0, stream>>>(localb, updb, incb, nullptr,
                                           bpp(blk, B_LN2S), bpp(blk, B_LN2B));
  }

  // ---- 6. output: where(is_t, local + LN(inc), 0) ----
  final_kernel<<<NN, 32, 0, stream>>>(localb, incb, prm(PI_FINLNS),
                                      prm(PI_FINLNB), istf, out);
}
